// NeuralSDEGenerator_11330123727258
// MI455X (gfx1250) — matrix operand order repaired and checked by reading, not compiled, not run on hardware
//
#include <hip/hip_runtime.h>
#include <math.h>

// ---------------------------------------------------------------------------
// NeuralSDE generator for MI455X (gfx1250): wave32 + v_wmma_f32_16x16x32_f16.
// Each block = 128 threads = 4 waves; each wave integrates 16 batch rows
// through all 99 Euler-Maruyama steps. All MLP weights are packed once into
// LDS in WMMA B-fragment order; activations live in per-wave LDS tiles; the
// f32 state x lives in registers of lanes 0..15.
// Round 2: branch-free tanh (hw v_tanh_f32 if available), A-fragment hoisting,
// fully unrolled WMMA tile loops.
// ---------------------------------------------------------------------------

typedef __attribute__((ext_vector_type(16))) _Float16 v16h;
typedef __attribute__((ext_vector_type(8)))  _Float16 v8h;
typedef __attribute__((ext_vector_type(8)))  float    v8f;

#define SDE_BATCH   65536
#define SDE_STEPS   99      // dW steps; path has 100 entries
#define ROWS_PER_WAVE 16
#define WAVES_PER_BLK 4
#define ROWS_PER_BLK  (ROWS_PER_WAVE * WAVES_PER_BLK)   // 64
#define TPB           (32 * WAVES_PER_BLK)              // 128

// ---- branch-free tanh ------------------------------------------------------
// Prefer the gfx1250 hardware transcendental (co-executes with WMMA); else use
// tanh(x) = (e^2x - 1)/(e^2x + 1) via v_exp_f32/v_rcp_f32, clamped so the
// exponential cannot overflow (tanh saturates long before |x|=10).
__device__ __forceinline__ float fast_tanh(float x) {
#if __has_builtin(__builtin_amdgcn_tanhf)
  return __builtin_amdgcn_tanhf(x);
#elif __has_builtin(__builtin_amdgcn_tanh_f32)
  return __builtin_amdgcn_tanh_f32(x);
#else
  float xc = fminf(fmaxf(x, -10.0f), 10.0f);
  float t  = __builtin_amdgcn_exp2f(2.8853900817779268f * xc);  // e^(2x)
  return (t - 1.0f) * __builtin_amdgcn_rcpf(t + 1.0f);
#endif
}

// --- A fragment (16x32 f16, row-major source, lane&15 = M row) --------------
// Lanes 0-15 hold K = [h*8, h*8+8) and [16+h*8, ...) contiguously (ISA 7.12.2)
__device__ __forceinline__ v16h ldsA(const _Float16* base, int row, int ld,
                                     int kbase, int half) {
  const v8h* p0 = reinterpret_cast<const v8h*>(base + row * ld + kbase + half * 8);
  const v8h* p1 = reinterpret_cast<const v8h*>(base + row * ld + kbase + 16 + half * 8);
  v8h lo = *p0, hi = *p1;
  v16h a;
#pragma unroll
  for (int i = 0; i < 8; ++i) { a[i] = lo[i]; a[8 + i] = hi[i]; }
  return a;
}

// --- B fragment from pre-packed LDS tile (512 f16 per 32x16 tile) -----------
__device__ __forceinline__ v16h ldsB(const _Float16* tileBase, int lane) {
  const v8h* p = reinterpret_cast<const v8h*>(tileBase + lane * 16);
  v8h lo = p[0], hi = p[1];
  v16h b;
#pragma unroll
  for (int i = 0; i < 8; ++i) { b[i] = lo[i]; b[8 + i] = hi[i]; }
  return b;
}

__device__ __forceinline__ v8f wmma16(v16h a, v16h b, v8f c) {
  return __builtin_amdgcn_wmma_f32_16x16x32_f16(false, a, false, b,
                                                (short)0, c, false, false);
}

// Dense layer: out[16][o_ld](f16) = tanh(A[16][KCHUNKS*32] @ Bp + bias)
template <int KCHUNKS, int NTILES>
__device__ __forceinline__ void layer16(const _Float16* A, int a_ld,
                                        const _Float16* Bp, const float* bias,
                                        _Float16* outp, int o_ld, int lane) {
  const int colr = lane & 15, half = lane >> 4;
  v16h afrag[KCHUNKS];
#pragma unroll
  for (int kc = 0; kc < KCHUNKS; ++kc)
    afrag[kc] = ldsA(A, colr, a_ld, kc * 32, half);
#pragma unroll
  for (int nt = 0; nt < NTILES; ++nt) {
    v8f acc = {};
#pragma unroll
    for (int kc = 0; kc < KCHUNKS; ++kc)
      acc = wmma16(afrag[kc], ldsB(Bp + (nt * KCHUNKS + kc) * 512, lane), acc);
    float bb = bias[nt * 16 + colr];
#pragma unroll
    for (int r = 0; r < 8; ++r) {
      float v = fast_tanh(acc[r] + bb);
      outp[(r + 8 * half) * o_ld + nt * 16 + colr] = (_Float16)v;
    }
  }
}

// Final layer: f32 output, no activation
template <int KCHUNKS, int NTILES>
__device__ __forceinline__ void layer32(const _Float16* A, int a_ld,
                                        const _Float16* Bp, const float* bias,
                                        float* outp, int o_ld, int lane) {
  const int colr = lane & 15, half = lane >> 4;
  v16h afrag[KCHUNKS];
#pragma unroll
  for (int kc = 0; kc < KCHUNKS; ++kc)
    afrag[kc] = ldsA(A, colr, a_ld, kc * 32, half);
#pragma unroll
  for (int nt = 0; nt < NTILES; ++nt) {
    v8f acc = {};
#pragma unroll
    for (int kc = 0; kc < KCHUNKS; ++kc)
      acc = wmma16(afrag[kc], ldsB(Bp + (nt * KCHUNKS + kc) * 512, lane), acc);
    float bb = bias[nt * 16 + colr];
#pragma unroll
    for (int r = 0; r < 8; ++r)
      outp[(r + 8 * half) * o_ld + nt * 16 + colr] = acc[r] + bb;
  }
}

// Pack row-major f32 weight [src_rows][src_cols] into B-fragment-ordered f16
// tiles: dst[tile*512 + lane*16 + i], tile = nt*kchunks + kc,
// element (lane,i) = src[kc*32 + (lane>=16)*8 + (i&7) + (i>>3)*16][nt*16 + (lane&15)]
__device__ __forceinline__ void packB(_Float16* dst, const float* src,
                                      int src_rows, int src_cols,
                                      int kchunks, int ntiles,
                                      int tid, int nthreads) {
  int total = ntiles * kchunks * 512;
  for (int e = tid; e < total; e += nthreads) {
    int tile = e >> 9;
    int w = e & 511;
    int lane = w >> 4;
    int i = w & 15;
    int nt = tile / kchunks;
    int kc = tile - nt * kchunks;
    int col = nt * 16 + (lane & 15);
    int K = kc * 32 + (lane >> 4) * 8 + (i & 7) + (i >> 3) * 16;
    float v = 0.f;
    if (K < src_rows && col < src_cols) v = src[K * src_cols + col];
    dst[e] = (_Float16)v;
  }
}

__device__ __forceinline__ void copyBias(float* dst, const float* src,
                                         int padded, int valid,
                                         int tid, int nthreads) {
  for (int j = tid; j < padded; j += nthreads)
    dst[j] = (j < valid) ? src[j] : 0.f;
}

__global__ __launch_bounds__(TPB, 1)
void neural_sde_kernel(const float* __restrict__ noise,     // [B,8]
                       const float* __restrict__ dW,        // [99,B,8]
                       const float* __restrict__ w1d, const float* __restrict__ b1d,
                       const float* __restrict__ w2d, const float* __restrict__ b2d,
                       const float* __restrict__ w3d, const float* __restrict__ b3d,
                       const float* __restrict__ w1f, const float* __restrict__ b1f,
                       const float* __restrict__ w2f, const float* __restrict__ b2f,
                       const float* __restrict__ w3f, const float* __restrict__ b3f,
                       const float* __restrict__ iw1, const float* __restrict__ ib1,
                       const float* __restrict__ iw2, const float* __restrict__ ib2,
                       float* __restrict__ out)              // [B, 100*8]
{
  // ---- packed weights (B-fragment order, f16) ----
  __shared__ _Float16 sW1d[8 * 512];    // (9->32) x 128
  __shared__ _Float16 sW2d[32 * 512];   // 128 x 128
  __shared__ _Float16 sW3d[4 * 512];    // 128 x (8->16)
  __shared__ _Float16 sW1f[8 * 512];
  __shared__ _Float16 sW2f[32 * 512];
  __shared__ _Float16 sW3f[16 * 512];   // 128 x 64
  __shared__ float sB1d[128], sB2d[128], sB3d[16];
  __shared__ float sB1f[128], sB2f[128], sB3f[64];
  // ---- per-wave activation tiles ----
  __shared__ _Float16 sTx[WAVES_PER_BLK * 16 * 32];   // [t,x,pad] f16
  __shared__ _Float16 sH1[WAVES_PER_BLK * 16 * 128];
  __shared__ _Float16 sH2[WAVES_PER_BLK * 16 * 128];
  __shared__ float    sFG[WAVES_PER_BLK * 16 * 80];   // cols 0..7 f, 16..79 g

  const int tid  = threadIdx.x;
  const int lane = tid & 31;
  const int wv   = tid >> 5;
  const int blockRow0 = blockIdx.x * ROWS_PER_BLK;

  // ---- stage + pack weights into LDS ----
  packB(sW1d, w1d,   9, 128, 1, 8, tid, TPB);
  packB(sW2d, w2d, 128, 128, 4, 8, tid, TPB);
  packB(sW3d, w3d, 128,   8, 4, 1, tid, TPB);
  packB(sW1f, w1f,   9, 128, 1, 8, tid, TPB);
  packB(sW2f, w2f, 128, 128, 4, 8, tid, TPB);
  packB(sW3f, w3f, 128,  64, 4, 4, tid, TPB);
  copyBias(sB1d, b1d, 128, 128, tid, TPB);
  copyBias(sB2d, b2d, 128, 128, tid, TPB);
  copyBias(sB3d, b3d,  16,   8, tid, TPB);
  copyBias(sB1f, b1f, 128, 128, tid, TPB);
  copyBias(sB2f, b2f, 128, 128, tid, TPB);
  copyBias(sB3f, b3f,  64,  64, tid, TPB);
  __syncthreads();

  const float dt  = 1.0f / (float)SDE_STEPS;
  const float sdt = sqrtf(dt);

  // ---- initial condition (scalar, once): x0 = tanh(n@iw1+ib1)@iw2+ib2 ----
  float xreg[8];   // f32 master state, valid in lanes 0..15
  _Float16* txw = sTx + wv * (16 * 32);
  if (lane < 16) {
    const int row = lane;
    const size_t gb = (size_t)blockRow0 + wv * 16 + row;
    float nz[8];
#pragma unroll
    for (int k = 0; k < 8; ++k) nz[k] = noise[gb * 8 + k];
    float acc[8];
#pragma unroll
    for (int d = 0; d < 8; ++d) acc[d] = ib2[d];
    for (int j = 0; j < 128; ++j) {
      float h = ib1[j];
#pragma unroll
      for (int k = 0; k < 8; ++k) h += nz[k] * iw1[k * 128 + j];
      h = fast_tanh(h);
#pragma unroll
      for (int d = 0; d < 8; ++d) acc[d] += h * iw2[j * 8 + d];
    }
    _Float16* txrow = txw + row * 32;
    txrow[0] = (_Float16)0.0f;                 // t = 0
#pragma unroll
    for (int d = 0; d < 8; ++d) {
      xreg[d] = acc[d];
      txrow[1 + d] = (_Float16)acc[d];
    }
    for (int c = 9; c < 32; ++c) txrow[c] = (_Float16)0.0f;  // K padding
    float4* op = reinterpret_cast<float4*>(out + gb * 800);
    op[0] = make_float4(acc[0], acc[1], acc[2], acc[3]);
    op[1] = make_float4(acc[4], acc[5], acc[6], acc[7]);
  }

  _Float16* h1w = sH1 + wv * (16 * 128);
  _Float16* h2w = sH2 + wv * (16 * 128);
  float*    fgw = sFG + wv * (16 * 80);

  // ---- Euler-Maruyama scan (per-wave private buffers: no block barriers) ----
  for (int step = 0; step < SDE_STEPS; ++step) {
    // drift MLP: tx -> h1 -> h2 -> f (fg cols 0..15)
    layer16<1, 8>(txw, 32, sW1d, sB1d, h1w, 128, lane);
    layer16<4, 8>(h1w, 128, sW2d, sB2d, h2w, 128, lane);
    layer32<4, 1>(h2w, 128, sW3d, sB3d, fgw, 80, lane);
    // diffusion MLP: tx -> h1 -> h2 -> g (fg cols 16..79)
    layer16<1, 8>(txw, 32, sW1f, sB1f, h1w, 128, lane);
    layer16<4, 8>(h1w, 128, sW2f, sB2f, h2w, 128, lane);
    layer32<4, 4>(h2w, 128, sW3f, sB3f, fgw + 16, 80, lane);

    // state update: dx = f*dt + sqrt(dt) * (g[d,:] . dW)
    if (lane < 16) {
      const int row = lane;
      const size_t gb = (size_t)blockRow0 + wv * 16 + row;
      const float4* dwp = reinterpret_cast<const float4*>(
          dW + ((size_t)step * SDE_BATCH + gb) * 8);
      float4 wa = dwp[0], wb = dwp[1];
      float dwv[8] = {wa.x, wa.y, wa.z, wa.w, wb.x, wb.y, wb.z, wb.w};
      float xn[8];
#pragma unroll
      for (int d = 0; d < 8; ++d) {
        const float* grow = fgw + row * 80 + 16 + d * 8;
        float dot = 0.f;
#pragma unroll
        for (int n = 0; n < 8; ++n) dot += grow[n] * dwv[n];
        float v = xreg[d] + fgw[row * 80 + d] * dt + sdt * dot;
        xreg[d] = v;
        xn[d] = v;
      }
      _Float16* txrow = txw + row * 32;
      txrow[0] = (_Float16)((float)(step + 1) * dt);   // t for next step
#pragma unroll
      for (int d = 0; d < 8; ++d) txrow[1 + d] = (_Float16)xn[d];
      float4* op = reinterpret_cast<float4*>(out + gb * 800 + (size_t)(step + 1) * 8);
      op[0] = make_float4(xn[0], xn[1], xn[2], xn[3]);
      op[1] = make_float4(xn[4], xn[5], xn[6], xn[7]);
    }
  }
}

extern "C" void kernel_launch(void* const* d_in, const int* in_sizes, int n_in,
                              void* d_out, int out_size, void* d_ws, size_t ws_size,
                              hipStream_t stream) {
  (void)in_sizes; (void)n_in; (void)d_ws; (void)ws_size; (void)out_size;
  // setup_inputs() order: 0=x (unused by the math), 1=initial_noise, 2=dW,
  // 3..8 drift w/b, 9..14 diff w/b, 15..18 init w/b
  const float* noise = (const float*)d_in[1];
  const float* dW    = (const float*)d_in[2];
  const float* w1d = (const float*)d_in[3];
  const float* b1d = (const float*)d_in[4];
  const float* w2d = (const float*)d_in[5];
  const float* b2d = (const float*)d_in[6];
  const float* w3d = (const float*)d_in[7];
  const float* b3d = (const float*)d_in[8];
  const float* w1f = (const float*)d_in[9];
  const float* b1f = (const float*)d_in[10];
  const float* w2f = (const float*)d_in[11];
  const float* b2f = (const float*)d_in[12];
  const float* w3f = (const float*)d_in[13];
  const float* b3f = (const float*)d_in[14];
  const float* iw1 = (const float*)d_in[15];
  const float* ib1 = (const float*)d_in[16];
  const float* iw2 = (const float*)d_in[17];
  const float* ib2 = (const float*)d_in[18];
  float* out = (float*)d_out;

  dim3 grid(SDE_BATCH / ROWS_PER_BLK);   // 1024 blocks
  dim3 block(TPB);                       // 128 threads = 4 waves
  neural_sde_kernel<<<grid, block, 0, stream>>>(
      noise, dW, w1d, b1d, w2d, b2d, w3d, b3d,
      w1f, b1f, w2f, b2f, w3f, b3f, iw1, ib1, iw2, ib2, out);
}
